// MyGraphConvModel_29403346109121
// MI455X (gfx1250) — matrix operand order repaired and checked by reading, not compiled, not run on hardware
//
#include <hip/hip_runtime.h>
#include <math.h>

// ---------------------------------------------------------------------------
// MI455X (gfx1250) GraphConv pipeline, round 4.
// - Input features repacked once to zero-padded f16 so ALL conv layers use
//   the vector-load + packed-f16 + WMMA path (no scalar/guarded loads left).
// - WMMA 16x16x32 f16->f32; 5 degree-variants per 16-atom tile, selected per
//   atom in the f32 accumulator. Weights staged in LDS in B-fragment layout.
// ---------------------------------------------------------------------------

typedef __attribute__((ext_vector_type(16))) _Float16 v16h;
typedef __attribute__((ext_vector_type(8)))  _Float16 vh8;
typedef __attribute__((ext_vector_type(8)))  float    v8f;

#define F0_FEAT 75
#define MAXD 4
#define NDEG 5
#define ENC_NEGINF 0x007FFFFFu

__device__ __forceinline__ v8f wmma_f16(v16h a, v16h b, v8f c) {
  return __builtin_amdgcn_wmma_f32_16x16x32_f16(false, a, false, b, (short)0, c,
                                                false, false);
}

__device__ __forceinline__ float selu_f(float x) {
  const float scale = 1.0507009873554805f;
  const float alpha = 1.6732632423543772f;
  return x > 0.f ? scale * x : scale * alpha * (__expf(x) - 1.f);
}

__device__ __forceinline__ unsigned enc_f32(float f) {
  unsigned u = __float_as_uint(f);
  return (u >> 31) ? ~u : (u | 0x80000000u);
}
__device__ __forceinline__ float dec_f32(unsigned e) {
  unsigned u = (e >> 31) ? (e & 0x7FFFFFFFu) : ~e;
  return __uint_as_float(u);
}

// Build a 16x32 f16 A-fragment from a zero-padded f16 row.
__device__ __forceinline__ v16h a_from_f16(const _Float16* rp) {
  const vh8 lo = *(const vh8*)(rp);       // j = 0..7
  const vh8 hi = *(const vh8*)(rp + 16);  // j = 8..15
  return __builtin_shufflevector(lo, hi, 0, 1, 2, 3, 4, 5, 6, 7, 8, 9, 10, 11,
                                 12, 13, 14, 15);
}

// ---------------------------------------------------------------------------
// Repack raw f32 features [N,75] -> zero-padded f16 [N,96] (streaming).
// ---------------------------------------------------------------------------
__global__ void repack_kernel(const float* __restrict__ X,
                              _Float16* __restrict__ Y, int n) {
  for (int i = blockIdx.x * blockDim.x + threadIdx.x; i < n;
       i += gridDim.x * blockDim.x) {
    const int row = i / 96;
    const int col = i - row * 96;
    const int cc = col < F0_FEAT ? col : F0_FEAT - 1;
    float v = X[(size_t)row * F0_FEAT + cc];
    Y[i] = (_Float16)(col < F0_FEAT ? v : 0.f);
  }
}

// ---------------------------------------------------------------------------
// Graph conv layer. One wave = 16 atoms. Input f16, zero-padded, row stride
// SIN == KSTEPS*32 halfs. Output f16, zero-padded to SOUT.
// ---------------------------------------------------------------------------
template <int FIN, int FOUT, int KSTEPS, int STILES, int NTILES, int SIN,
          int SOUT>
__global__ __launch_bounds__(256) void graphconv_kernel(
    const _Float16* __restrict__ X, const float* __restrict__ Ws,
    const float* __restrict__ Wn, const float* __restrict__ Bb,
    const int* __restrict__ nbr, const int* __restrict__ deg,
    _Float16* __restrict__ Y, int n) {
  constexpr int NFRAG = NDEG * KSTEPS * NTILES * 2;
  __shared__ v16h ldsW[NFRAG * 32];

  {  // cooperative fill: clamped unconditional loads, value-select padding
    _Float16* lw = (_Float16*)ldsW;
    for (int e = threadIdx.x; e < NFRAG * 512; e += 256) {
      const int frag = e >> 9;
      const int r = e & 511;
      const int ln = r >> 4, j = r & 15;
      const int which = frag & 1;
      const int tmp = frag >> 1;
      const int d = tmp % NDEG;
      const int tmp2 = tmp / NDEG;
      const int ks = tmp2 % KSTEPS;
      const int nt = tmp2 / KSTEPS;
      const int col = nt * 16 + (ln & 15);
      const int k = ks * 32 + 8 * (ln >> 4) + (j < 8 ? j : j + 8);
      const int kc = k < FIN ? k : FIN - 1;
      const int cc = col < FOUT ? col : FOUT - 1;
      const float* Wsrc = which ? Wn : Ws;
      float v = Wsrc[(d * FIN + kc) * FOUT + cc];
      v = (k < FIN && col < FOUT) ? v : 0.f;
      lw[e] = (_Float16)v;
    }
  }
  __syncthreads();

  const int lane = threadIdx.x & 31;
  const int wave = (blockIdx.x * blockDim.x + threadIdx.x) >> 5;
  const int rowBase = wave << 4;
  if (rowBase >= n) return;  // wave-uniform (after barrier)
  const int g = lane >> 4;
  const int arowRaw = rowBase + (lane & 15);
  const int arow = arowRaw < n ? arowRaw : n - 1;  // clamped: loads always safe

  const int mdeg = deg[arow];
  int mnbr[MAXD];
  _Float16 hm[MAXD];
#pragma unroll
  for (int t = 0; t < MAXD; ++t) {
    mnbr[t] = nbr[arow * MAXD + t];
    hm[t] = (_Float16)(t < mdeg ? 1.f : 0.f);
  }

  v16h aself[KSTEPS], anbr[KSTEPS];
#pragma unroll
  for (int ks = 0; ks < KSTEPS; ++ks) {
    const _Float16* rp = X + (size_t)arow * SIN + ks * 32 + 8 * g;
    aself[ks] = a_from_f16(rp);
    vh8 nlo = {}, nhi = {};
#pragma unroll
    for (int t = 0; t < MAXD; ++t) {
      const _Float16* np = X + (size_t)mnbr[t] * SIN + ks * 32 + 8 * g;
      nlo += hm[t] * *(const vh8*)(np);
      nhi += hm[t] * *(const vh8*)(np + 16);
    }
    anbr[ks] = __builtin_shufflevector(nlo, nhi, 0, 1, 2, 3, 4, 5, 6, 7, 8, 9,
                                       10, 11, 12, 13, 14, 15);
  }

  int dout[8];
#pragma unroll
  for (int r = 0; r < 8; ++r) {
    const int ar = rowBase + r + 8 * g;
    dout[r] = deg[ar < n ? ar : n - 1];
  }

#pragma unroll
  for (int nt = 0; nt < STILES; ++nt) {
    const int col = nt * 16 + (lane & 15);
    if (nt < NTILES) {
      v8f zero = {};
      v8f acc[NDEG];
#pragma unroll
      for (int d = 0; d < NDEG; ++d) acc[d] = zero;
#pragma unroll
      for (int ks = 0; ks < KSTEPS; ++ks) {
#pragma unroll
        for (int d = 0; d < NDEG; ++d) {
          const int fb = ((nt * KSTEPS + ks) * NDEG + d) * 2;
          const v16h bs = ldsW[fb * 32 + lane];
          const v16h bn = ldsW[(fb + 1) * 32 + lane];
          acc[d] = wmma_f16(aself[ks], bs, acc[d]);
          acc[d] = wmma_f16(anbr[ks], bn, acc[d]);
        }
      }
      const int cc = col < FOUT ? col : FOUT - 1;
#pragma unroll
      for (int r = 0; r < 8; ++r) {
        const int ar = rowBase + r + 8 * g;
        const int d = dout[r];
        float v = acc[0][r];
        v = (d == 1) ? acc[1][r] : v;
        v = (d == 2) ? acc[2][r] : v;
        v = (d == 3) ? acc[3][r] : v;
        v = (d == 4) ? acc[4][r] : v;
        v += Bb[d * FOUT + cc];
        v = (col < FOUT) ? selu_f(v) : 0.f;
        if (ar < n) Y[(size_t)ar * SOUT + col] = (_Float16)v;
      }
    } else {
#pragma unroll
      for (int r = 0; r < 8; ++r) {
        const int ar = rowBase + r + 8 * g;
        if (ar < n) Y[(size_t)ar * SOUT + col] = (_Float16)0.f;
      }
    }
  }
}

// ---------------------------------------------------------------------------
// Wave GEMM: C[M x SOUT] = act(A[M x K] @ W[K x NCOLS] + bias).
// A is f16, row stride KSTEPS*32 halfs, zero-padded. OUTH: f16 vs f32 output.
// ---------------------------------------------------------------------------
enum { ACT_NONE = 0, ACT_SOFTPAIR = 1, ACT_SIGMOID = 2 };

template <int K, int NCOLS, int KSTEPS, int NTILES, int SOUT, int ACT,
          bool OUTH>
__global__ __launch_bounds__(256) void gemm16_kernel(
    const _Float16* __restrict__ A, const float* __restrict__ W,
    const float* __restrict__ bias, void* __restrict__ Cv, int M) {
  constexpr int NFRAG = KSTEPS * NTILES;
  __shared__ v16h ldsW[NFRAG * 32];
  {
    _Float16* lw = (_Float16*)ldsW;
    for (int e = threadIdx.x; e < NFRAG * 512; e += 256) {
      const int frag = e >> 9;
      const int r = e & 511;
      const int ln = r >> 4, j = r & 15;
      const int ks = frag % KSTEPS;
      const int nt = frag / KSTEPS;
      const int col = nt * 16 + (ln & 15);
      const int k = ks * 32 + 8 * (ln >> 4) + (j < 8 ? j : j + 8);
      const int kc = k < K ? k : K - 1;
      const int cc = col < NCOLS ? col : NCOLS - 1;
      float v = W[kc * NCOLS + cc];
      v = (k < K && col < NCOLS) ? v : 0.f;
      lw[e] = (_Float16)v;
    }
  }
  __syncthreads();

  const int lane = threadIdx.x & 31;
  const int wave = (blockIdx.x * blockDim.x + threadIdx.x) >> 5;
  const int rowBase = wave << 4;
  if (rowBase >= M) return;
  const int g = lane >> 4;
  const int arowRaw = rowBase + (lane & 15);
  const int arow = arowRaw < M ? arowRaw : M - 1;
  constexpr int SA = KSTEPS * 32;

  v16h afrag[KSTEPS];
#pragma unroll
  for (int ks = 0; ks < KSTEPS; ++ks)
    afrag[ks] = a_from_f16(A + (size_t)arow * SA + ks * 32 + 8 * g);

  float* Cf = (float*)Cv;
  _Float16* Ch = (_Float16*)Cv;

#pragma unroll
  for (int nt = 0; nt < NTILES; ++nt) {
    const int col = nt * 16 + (lane & 15);
    v8f acc = {};
#pragma unroll
    for (int ks = 0; ks < KSTEPS; ++ks)
      acc = wmma_f16(afrag[ks], ldsW[(nt * KSTEPS + ks) * 32 + lane], acc);

    const float bcol = bias[col < NCOLS ? col : NCOLS - 1];
#pragma unroll
    for (int r = 0; r < 8; ++r) {
      const int ar = rowBase + r + 8 * g;
      float v = acc[r] + bcol;
      if (ACT == ACT_SOFTPAIR) {
        const float o = __shfl_xor(v, 1, 32);
        const float mx = fmaxf(v, o);
        const float e = __expf(v - mx);
        const float eo = __expf(o - mx);
        v = e / (e + eo);
      } else if (ACT == ACT_SIGMOID) {
        v = 1.f / (1.f + __expf(-v));
      }
      v = (col < NCOLS) ? v : 0.f;
      if (ar < M && col < SOUT) {
        if (OUTH)
          Ch[(size_t)ar * SOUT + col] = (_Float16)v;
        else
          Cf[(size_t)ar * SOUT + col] = v;
      }
    }
  }
}

// ---------------------------------------------------------------------------
// Readout: segment sum + max over sorted membership. x4 f16, row stride 48.
// ---------------------------------------------------------------------------
__global__ void readout_init(float* __restrict__ ssum,
                             unsigned* __restrict__ smax, int n) {
  for (int i = blockIdx.x * blockDim.x + threadIdx.x; i < n;
       i += gridDim.x * blockDim.x) {
    ssum[i] = 0.f;
    smax[i] = ENC_NEGINF;
  }
}

__global__ void readout_scatter(const _Float16* __restrict__ x,
                                const int* __restrict__ mem,
                                float* __restrict__ ssum,
                                unsigned* __restrict__ smax, int n) {
  for (int i = blockIdx.x * blockDim.x + threadIdx.x; i < n;
       i += gridDim.x * blockDim.x) {
    const int atom = i / 36;
    const int f = i - atom * 36;
    const int gidx = mem[atom];
    const float v = (float)x[(size_t)atom * 48 + f];
    atomicAdd(&ssum[gidx * 36 + f], v);
    atomicMax(&smax[gidx * 36 + f], enc_f32(v));
  }
}

// r: f16, row stride 96 (0..35 selu(sum), 36..71 selu(max), 72..95 zero)
__global__ void readout_finish(const float* __restrict__ ssum,
                               const unsigned* __restrict__ smax,
                               _Float16* __restrict__ r, int n) {
  for (int i = blockIdx.x * blockDim.x + threadIdx.x; i < n;
       i += gridDim.x * blockDim.x) {
    const int gidx = i / 96;
    const int col = i - gidx * 96;
    float v = 0.f;
    if (col < 36) {
      v = selu_f(ssum[gidx * 36 + col]);
    } else if (col < 72) {
      const unsigned e = smax[gidx * 36 + (col - 36)];
      const float mv = (e == ENC_NEGINF) ? 0.f : dec_f32(e);
      v = selu_f(mv);
    }
    r[i] = (_Float16)v;
  }
}

// ---------------------------------------------------------------------------
// Batch norm (biased variance, eps=1e-5) + relu; S = row stride >= F.
// ---------------------------------------------------------------------------
__global__ void zero_kernel(float* __restrict__ p, int n) {
  for (int i = blockIdx.x * blockDim.x + threadIdx.x; i < n;
       i += gridDim.x * blockDim.x)
    p[i] = 0.f;
}

__global__ void bn_stats_kernel(const float* __restrict__ z,
                                float* __restrict__ stats, int rows, int F,
                                int S) {
  const int n = rows * S;
  for (int i = blockIdx.x * blockDim.x + threadIdx.x; i < n;
       i += gridDim.x * blockDim.x) {
    const int c = i % S;
    if (c < F) {
      const float v = z[i];
      atomicAdd(&stats[c], v);
      atomicAdd(&stats[F + c], v * v);
    }
  }
}

__global__ void bn_apply_kernel(const float* __restrict__ z,
                                const float* __restrict__ stats,
                                const float* __restrict__ gamma,
                                const float* __restrict__ beta,
                                _Float16* __restrict__ h, int rows, int F,
                                int S) {
  const int n = rows * S;
  const float inv = 1.f / (float)rows;
  for (int i = blockIdx.x * blockDim.x + threadIdx.x; i < n;
       i += gridDim.x * blockDim.x) {
    const int c = i % S;
    const int cc = c < F ? c : F - 1;
    const float m = stats[cc] * inv;
    const float var = stats[F + cc] * inv - m * m;
    float v = (z[i] - m) * rsqrtf(var + 1e-5f) * gamma[cc] + beta[cc];
    v = fmaxf(v, 0.f);
    h[i] = (_Float16)((c < F) ? v : 0.f);
  }
}

// ---------------------------------------------------------------------------
extern "C" void kernel_launch(void* const* d_in, const int* in_sizes, int n_in,
                              void* d_out, int out_size, void* d_ws,
                              size_t ws_size, hipStream_t stream) {
  (void)n_in;
  (void)ws_size;
  const float* atom = (const float*)d_in[0];
  const float* Ws1 = (const float*)d_in[1];
  const float* Wn1 = (const float*)d_in[2];
  const float* b1 = (const float*)d_in[3];
  const float* Ws2 = (const float*)d_in[4];
  const float* Wn2 = (const float*)d_in[5];
  const float* b2 = (const float*)d_in[6];
  const float* Ws3 = (const float*)d_in[7];
  const float* Wn3 = (const float*)d_in[8];
  const float* b3 = (const float*)d_in[9];
  const float* Ws4 = (const float*)d_in[10];
  const float* Wn4 = (const float*)d_in[11];
  const float* b4 = (const float*)d_in[12];
  const float* Wd2 = (const float*)d_in[13];
  const float* bd2 = (const float*)d_in[14];
  const float* Wl1 = (const float*)d_in[15];
  const float* bl1 = (const float*)d_in[16];
  const float* g1 = (const float*)d_in[17];
  const float* be1 = (const float*)d_in[18];
  const float* Wl2 = (const float*)d_in[19];
  const float* bl2 = (const float*)d_in[20];
  const float* g2 = (const float*)d_in[21];
  const float* be2 = (const float*)d_in[22];
  const float* Wl3 = (const float*)d_in[23];
  const float* bl3 = (const float*)d_in[24];
  const int* nbr = (const int*)d_in[25];
  const int* deg = (const int*)d_in[26];
  const int* mem = (const int*)d_in[27];

  const int N = in_sizes[0] / F0_FEAT;  // 400000
  const int Bsz = out_size / 138;       // 16384

  // Workspace. f16 activations: x0 stride 96, xA 32, xB 48, r 96, p 32, h 96.
  char* w = (char*)d_ws;
  size_t off = 0;
  auto carve = [&](size_t bytes) -> void* {
    void* p = w + off;
    off += (bytes + 255) & ~(size_t)255;
    return p;
  };
  _Float16* x0 = (_Float16*)carve((size_t)N * 96 * sizeof(_Float16));
  _Float16* xA = (_Float16*)carve((size_t)N * 32 * sizeof(_Float16));
  _Float16* xB = (_Float16*)carve((size_t)N * 48 * sizeof(_Float16));
  float* ssum = (float*)carve((size_t)Bsz * 36 * sizeof(float));
  unsigned* smax = (unsigned*)carve((size_t)Bsz * 36 * sizeof(unsigned));
  _Float16* rbuf = (_Float16*)carve((size_t)Bsz * 96 * sizeof(_Float16));
  _Float16* pbuf = (_Float16*)carve((size_t)Bsz * 32 * sizeof(_Float16));
  float* zbuf = (float*)carve((size_t)Bsz * 96 * sizeof(float));
  _Float16* hbuf = (_Float16*)carve((size_t)Bsz * 96 * sizeof(_Float16));
  float* stats = (float*)carve(2 * 96 * sizeof(float));

  const dim3 blk(256);
  const int convBlocks = (((N + 15) / 16) * 32 + 255) / 256;

  // repack raw features -> padded f16 [N,96]
  repack_kernel<<<2048, blk, 0, stream>>>(atom, x0, N * 96);

  // <FIN,FOUT,KSTEPS,STILES,NTILES,SIN,SOUT>
  graphconv_kernel<75, 15, 3, 2, 1, 96, 32>
      <<<convBlocks, blk, 0, stream>>>(x0, Ws1, Wn1, b1, nbr, deg, xA, N);
  graphconv_kernel<15, 20, 1, 2, 2, 32, 32>
      <<<convBlocks, blk, 0, stream>>>(xA, Ws2, Wn2, b2, nbr, deg, xB, N);
  graphconv_kernel<20, 27, 1, 2, 2, 32, 32>
      <<<convBlocks, blk, 0, stream>>>(xB, Ws3, Wn3, b3, nbr, deg, xA, N);
  graphconv_kernel<27, 36, 1, 3, 3, 32, 48>
      <<<convBlocks, blk, 0, stream>>>(xA, Ws4, Wn4, b4, nbr, deg, xB, N);

  // GraphGather readout -> r [B,96] f16 (72 valid + pad)
  const int nbf = Bsz * 36;
  readout_init<<<(nbf + 255) / 256, blk, 0, stream>>>(ssum, smax, nbf);
  readout_scatter<<<2048, blk, 0, stream>>>(xB, mem, ssum, smax, N * 36);
  readout_finish<<<(Bsz * 96 + 255) / 256, blk, 0, stream>>>(ssum, smax, rbuf,
                                                             Bsz * 96);

  const int gw = (((Bsz + 15) / 16) * 32 + 255) / 256;

  // logits + pairwise softmax -> p [B,32] f16 (24 valid + pad)
  gemm16_kernel<72, 24, 3, 2, 32, ACT_SOFTPAIR, true>
      <<<gw, blk, 0, stream>>>(rbuf, Wd2, bd2, pbuf, Bsz);

  // dense 1: p @ Wl1 -> bn -> relu -> h1 [B,96] f16
  gemm16_kernel<24, 96, 1, 6, 96, ACT_NONE, false>
      <<<gw, blk, 0, stream>>>(pbuf, Wl1, bl1, zbuf, Bsz);
  zero_kernel<<<1, 256, 0, stream>>>(stats, 192);
  bn_stats_kernel<<<1024, blk, 0, stream>>>(zbuf, stats, Bsz, 96, 96);
  bn_apply_kernel<<<(Bsz * 96 + 255) / 256, blk, 0, stream>>>(
      zbuf, stats, g1, be1, hbuf, Bsz, 96, 96);

  // dense 2: h1 @ Wl2 -> bn -> relu -> h2 [B,64] f16 (63 valid + pad)
  gemm16_kernel<96, 63, 3, 4, 64, ACT_NONE, false>
      <<<gw, blk, 0, stream>>>(hbuf, Wl2, bl2, zbuf, Bsz);
  zero_kernel<<<1, 256, 0, stream>>>(stats, 126);
  bn_stats_kernel<<<1024, blk, 0, stream>>>(zbuf, stats, Bsz, 63, 64);
  bn_apply_kernel<<<(Bsz * 64 + 255) / 256, blk, 0, stream>>>(
      zbuf, stats, g2, be2, hbuf, Bsz, 63, 64);

  // dense 3 + sigmoid -> d_out [B,138] f32
  gemm16_kernel<63, 138, 2, 9, 138, ACT_SIGMOID, false>
      <<<gw, blk, 0, stream>>>(hbuf, Wl3, bl3, d_out, Bsz);
}
